// RQVAEQuantizer_4140348473633
// MI455X (gfx1250) — matrix operand order repaired and checked by reading, not compiled, not run on hardware
//
#include <hip/hip_runtime.h>
#include <hip/hip_bf16.h>

typedef __attribute__((ext_vector_type(16))) _Float16 v16h;
typedef __attribute__((ext_vector_type(8)))  float    v8f;
typedef unsigned int u32x4 __attribute__((ext_vector_type(4)));
typedef int          i32x8 __attribute__((ext_vector_type(8)));
typedef int          i32x4 __attribute__((ext_vector_type(4)));

#define VQ_DECAY 0.99f
#define VQ_OMD   0.01f
#define VQ_EPS   1e-5f

// ---------------------------------------------------------------------------
// Kernel 0: init workspace — zero accumulators, fp16 codebook, 0.5*||e||^2
// ---------------------------------------------------------------------------
__global__ void vq_init_kernel(const float* __restrict__ emb,
                               float* __restrict__ wsES,
                               float* __restrict__ wsCounts,
                               float* __restrict__ wsHN,
                               _Float16* __restrict__ wsEH,
                               float* __restrict__ wsCommit,
                               int K)
{
    int gid    = blockIdx.x * blockDim.x + threadIdx.x;
    int stride = gridDim.x * blockDim.x;
    int total  = K * 32;
    for (int i = gid; i < total; i += stride) {
        float e = emb[i];
        wsES[i] = 0.0f;
        wsEH[i] = (_Float16)e;
    }
    if (gid < K) {
        wsCounts[gid] = 0.0f;
        const float* er = emb + (size_t)gid * 32;
        float s = 0.0f;
#pragma unroll
        for (int d = 0; d < 32; ++d) { float v = er[d]; s += v * v; }
        wsHN[gid] = 0.5f * s;
    }
    if (gid == 0) wsCommit[0] = 0.0f;
}

// ---------------------------------------------------------------------------
// Kernel 1: WMMA nearest-code search + gather + scatter-add accumulation
//   block = 256 threads (8 waves); each wave owns 32 rows -> 256 rows/block
// ---------------------------------------------------------------------------
__global__ __launch_bounds__(256) void vq_assign_kernel(
    const float* __restrict__ z,
    const float* __restrict__ emb,
    const _Float16* __restrict__ wsEH,
    const float* __restrict__ wsHN,
    float* __restrict__ wsES,
    float* __restrict__ wsCounts,
    float* __restrict__ wsCommit,
    float* __restrict__ out,
    int N, int K)
{
    extern __shared__ char smem[];
    _Float16* sEH  = (_Float16*)smem;                               // K*32 halfs
    float*    sHN  = (float*)(smem + (size_t)K * 64);               // K floats
    int*      sIdx = (int*)  (smem + (size_t)K * 68);               // 256 ints
    float*    sRed = (float*)(smem + (size_t)K * 68 + 1024);        // 256 floats

    const int t = threadIdx.x;

    // LDS byte offsets = low 32 bits of the generic shared-space addresses
    const unsigned ldsEHBase = (unsigned)(uintptr_t)(void*)sEH;
    const unsigned ldsHNBase = (unsigned)(uintptr_t)(void*)sHN;

    // ---- fp16 codebook (64KB) -> LDS via Tensor Data Mover ----
#if __has_builtin(__builtin_amdgcn_tensor_load_to_lds)
    if (t == 0) {
        unsigned long long ga = (unsigned long long)(const void*)wsEH;
        // D# group0: count=1 | lds_addr | global_addr[56:0] | type=2
        u32x4 g0 = { 1u,
                     ldsEHBase,
                     (unsigned)ga,
                     (unsigned)((ga >> 32) & 0x1FFFFFFull) | (2u << 30) };
        // D# group1: data_size=3(8B); tensor_dim0=8192; tensor_dim1=1;
        //            tile_dim0=8192; tile_dim1=1; dim0_stride=8192
        i32x8 g1 = { 0x00030000,    // wg_mask=0, data_size=3
                     0x20000000,    // tensor_dim0[15:0]=8192 in bits[31:16]
                     0x00010000,    // tensor_dim0 hi=0, tensor_dim1 lo=1
                     0x20000000,    // tensor_dim1 hi=0, tile_dim0=8192
                     0x00000001,    // tile_dim1=1, tile_dim2=0
                     0x00002000,    // tensor_dim0_stride lo = 8192
                     0, 0 };
        i32x4 gz = { 0, 0, 0, 0 };
#if __has_include(<hip/amd_detail/amd_gfx1250_TDM.h>)
        i32x8 gz8 = { 0, 0, 0, 0, 0, 0, 0, 0 };
        __builtin_amdgcn_tensor_load_to_lds(g0, g1, gz, gz, gz8, 0);
#else
        __builtin_amdgcn_tensor_load_to_lds(g0, g1, gz, gz, 0);
#endif
        __builtin_amdgcn_s_wait_tensorcnt(0);
    }
#else
    {
        const uint4* gEH = (const uint4*)wsEH;
        const int n16 = (K * 64) / 16;
        for (int i = t; i < n16; i += 256) {
            unsigned dst = ldsEHBase + (unsigned)i * 16u;
            const uint4* src = gEH + i;
            asm volatile("global_load_async_to_lds_b128 %0, %1, off"
                         :: "v"(dst), "v"(src) : "memory");
        }
    }
#endif

    // ---- 0.5*||e||^2 table (4KB) -> LDS via GLOBAL_LOAD_ASYNC_TO_LDS ----
    {
        const uint4* gHN = (const uint4*)wsHN;
        const int nh = (K * 4) / 16;
        for (int i = t; i < nh; i += 256) {
            unsigned dst = ldsHNBase + (unsigned)i * 16u;
            const uint4* src = gHN + i;
            asm volatile("global_load_async_to_lds_b128 %0, %1, off"
                         :: "v"(dst), "v"(src) : "memory");
        }
        asm volatile("s_wait_asynccnt 0x0" ::: "memory");
    }
    __syncthreads();

    const int lane = t & 31;
    const int wave = t >> 5;
    const int h    = lane >> 4;      // 16-lane half selector
    const int l16  = lane & 15;
    const int rowBase = blockIdx.x * 256;
    const int row0    = rowBase + wave * 32 + l16;     // A tile 0: rows +0..15
    const int row1    = row0 + 16;                     // A tile 1: rows +16..31

    // ---- build two A operands: 16x32 f16 (lanes0-15 hold K=0-7,16-23) ----
    v16h a0, a1;
    {
        const float* zr = z + (size_t)row0 * 32 + 8 * h;
        float4 f0 = ((const float4*)zr)[0];
        float4 f1 = ((const float4*)zr)[1];
        float4 f2 = ((const float4*)(zr + 16))[0];
        float4 f3 = ((const float4*)(zr + 16))[1];
        a0[0]=(_Float16)f0.x; a0[1]=(_Float16)f0.y; a0[2]=(_Float16)f0.z; a0[3]=(_Float16)f0.w;
        a0[4]=(_Float16)f1.x; a0[5]=(_Float16)f1.y; a0[6]=(_Float16)f1.z; a0[7]=(_Float16)f1.w;
        a0[8]=(_Float16)f2.x; a0[9]=(_Float16)f2.y; a0[10]=(_Float16)f2.z; a0[11]=(_Float16)f2.w;
        a0[12]=(_Float16)f3.x; a0[13]=(_Float16)f3.y; a0[14]=(_Float16)f3.z; a0[15]=(_Float16)f3.w;
    }
    {
        const float* zr = z + (size_t)row1 * 32 + 8 * h;
        float4 f0 = ((const float4*)zr)[0];
        float4 f1 = ((const float4*)zr)[1];
        float4 f2 = ((const float4*)(zr + 16))[0];
        float4 f3 = ((const float4*)(zr + 16))[1];
        a1[0]=(_Float16)f0.x; a1[1]=(_Float16)f0.y; a1[2]=(_Float16)f0.z; a1[3]=(_Float16)f0.w;
        a1[4]=(_Float16)f1.x; a1[5]=(_Float16)f1.y; a1[6]=(_Float16)f1.z; a1[7]=(_Float16)f1.w;
        a1[8]=(_Float16)f2.x; a1[9]=(_Float16)f2.y; a1[10]=(_Float16)f2.z; a1[11]=(_Float16)f2.w;
        a1[12]=(_Float16)f3.x; a1[13]=(_Float16)f3.y; a1[14]=(_Float16)f3.z; a1[15]=(_Float16)f3.w;
    }

    float bestVal0[8], bestVal1[8];
    int   bestIdx0[8], bestIdx1[8];
#pragma unroll
    for (int v = 0; v < 8; ++v) {
        bestVal0[v] = -3.4e38f; bestIdx0[v] = 0;
        bestVal1[v] = -3.4e38f; bestIdx1[v] = 0;
    }

    // ---- scan code tiles: one B load feeds two WMMAs (K-dim 32 == D) ----
    const int nkt = K >> 4;
#pragma unroll 2
    for (int kt = 0; kt < nkt; ++kt) {
        const int k = (kt << 4) + l16;
        typedef __attribute__((ext_vector_type(8))) _Float16 v8h_;
        v8h_ bl = *(const v8h_*)(sEH + (size_t)k * 32 + 8 * h);
        v8h_ bh = *(const v8h_*)(sEH + (size_t)k * 32 + 16 + 8 * h);
        v16h b = __builtin_shufflevector(bl, bh, 0,1,2,3,4,5,6,7,8,9,10,11,12,13,14,15);
        const float negq = -sHN[k];           // fold -0.5*||e||^2 into C
        v8f c0, c1;
#pragma unroll
        for (int v = 0; v < 8; ++v) { c0[v] = negq; c1[v] = negq; }
        c0 = __builtin_amdgcn_wmma_f32_16x16x32_f16(false, a0, false, b,
                                                    (short)0, c0, false, false);
        c1 = __builtin_amdgcn_wmma_f32_16x16x32_f16(false, a1, false, b,
                                                    (short)0, c1, false, false);
#pragma unroll
        for (int v = 0; v < 8; ++v) {
            float s0 = c0[v];
            if (s0 > bestVal0[v]) { bestVal0[v] = s0; bestIdx0[v] = k; }
            float s1 = c1[v];
            if (s1 > bestVal1[v]) { bestVal1[v] = s1; bestIdx1[v] = k; }
        }
    }

    // ---- argmax reduce across the 16 lanes of each half (wave32 shuffles) ----
#pragma unroll
    for (int off = 1; off < 16; off <<= 1) {
#pragma unroll
        for (int v = 0; v < 8; ++v) {
            float ov = __shfl_xor(bestVal0[v], off, 32);
            int   oi = __shfl_xor(bestIdx0[v], off, 32);
            if (ov > bestVal0[v] || (ov == bestVal0[v] && oi < bestIdx0[v])) {
                bestVal0[v] = ov; bestIdx0[v] = oi;
            }
            ov = __shfl_xor(bestVal1[v], off, 32);
            oi = __shfl_xor(bestIdx1[v], off, 32);
            if (ov > bestVal1[v] || (ov == bestVal1[v] && oi < bestIdx1[v])) {
                bestVal1[v] = ov; bestIdx1[v] = oi;
            }
        }
    }
    if (l16 == 0) {
#pragma unroll
        for (int v = 0; v < 8; ++v) {
            sIdx[wave * 32 + h * 8 + v]      = bestIdx0[v];
            sIdx[wave * 32 + 16 + h * 8 + v] = bestIdx1[v];
        }
    }
    __syncthreads();

    // ---- phase 2: one thread per row, full 32 dims, float4 I/O ----
    const int idx  = sIdx[t];
    const int grow = rowBase + t;

    const float4* z4 = (const float4*)(z   + (size_t)grow * 32);
    const float4* e4 = (const float4*)(emb + (size_t)idx  * 32);
    float4*       q4 = (float4*)(out + (size_t)grow * 32);
    float*     esrow = wsES + (size_t)idx * 32;

    float csum = 0.0f;
#pragma unroll
    for (int j = 0; j < 8; ++j) {
        float4 zv = z4[j];
        float4 ev = e4[j];
        float4 o;
        o.x = zv.x + (ev.x - zv.x);
        o.y = zv.y + (ev.y - zv.y);
        o.z = zv.z + (ev.z - zv.z);
        o.w = zv.w + (ev.w - zv.w);
        q4[j] = o;
        float dx = zv.x - ev.x, dy = zv.y - ev.y;
        float dz = zv.z - ev.z, dw = zv.w - ev.w;
        csum += dx * dx + dy * dy + dz * dz + dw * dw;
        atomicAdd(&esrow[j * 4 + 0], zv.x);
        atomicAdd(&esrow[j * 4 + 1], zv.y);
        atomicAdd(&esrow[j * 4 + 2], zv.z);
        atomicAdd(&esrow[j * 4 + 3], zv.w);
    }
    out[(size_t)N * 32 + 2 + grow] = (float)idx;       // encoding_indices
    atomicAdd(&wsCounts[idx], 1.0f);

    // block-reduce commitment partial
    sRed[t] = csum;
    __syncthreads();
    for (int off = 128; off > 0; off >>= 1) {
        if (t < off) sRed[t] += sRed[t + off];
        __syncthreads();
    }
    if (t == 0) atomicAdd(wsCommit, sRed[0]);
}

// ---------------------------------------------------------------------------
// Kernel 2: EMA finalize — cluster sizes, Laplace smoothing, new codebook
// ---------------------------------------------------------------------------
__global__ __launch_bounds__(1024) void vq_finalize_kernel(
    const float* __restrict__ cs,
    const float* __restrict__ eavg,
    const float* __restrict__ wsES,
    const float* __restrict__ wsCounts,
    const float* __restrict__ wsCommit,
    float* __restrict__ out,
    int N, int K)
{
    __shared__ float sred[1024];
    const int t = threadIdx.x;

    float cnt = (t < K) ? wsCounts[t] : 0.0f;
    float csn = (t < K) ? (VQ_DECAY * cs[t] + VQ_OMD * cnt) : 0.0f;

    sred[t] = csn;
    __syncthreads();
    for (int off = 512; off > 0; off >>= 1) {
        if (t < off) sred[t] += sred[t + off];
        __syncthreads();
    }
    float n = sred[0];
    __syncthreads();

    sred[t] = (t < K && cnt == 0.0f) ? 1.0f : 0.0f;
    __syncthreads();
    for (int off = 512; off > 0; off >>= 1) {
        if (t < off) sred[t] += sred[t + off];
        __syncthreads();
    }
    float unused = sred[0];

    const long long base = (long long)N * 32;
    const long long IDXo = base + 2;
    const long long UN   = IDXo + N;
    const long long EMB  = UN + 1;
    const long long CSo  = EMB + (long long)K * 32;
    const long long EAVG = CSo + K;

    if (t < K) {
        out[CSo + t] = csn;
        float norm = n * (csn + VQ_EPS) / (n + (float)K * VQ_EPS);
#pragma unroll
        for (int d = 0; d < 32; ++d) {
            float ea = VQ_DECAY * eavg[(size_t)t * 32 + d] + VQ_OMD * wsES[(size_t)t * 32 + d];
            out[EAVG + (long long)t * 32 + d] = ea;
            out[EMB  + (long long)t * 32 + d] = ea / norm;
        }
    }
    if (t == 0) {
        out[base]     = 0.0f;                                  // codebook_loss
        out[base + 1] = wsCommit[0] / ((float)N * 32.0f);      // commitment_loss
        out[UN]       = unused;                                // unused_codes
    }
}

// ---------------------------------------------------------------------------
extern "C" void kernel_launch(void* const* d_in, const int* in_sizes, int n_in,
                              void* d_out, int out_size, void* d_ws, size_t ws_size,
                              hipStream_t stream)
{
    const float* z    = (const float*)d_in[0];   // [N,32]
    const float* emb  = (const float*)d_in[1];   // [K,32]
    const float* eavg = (const float*)d_in[2];   // [K,32]
    const float* cs   = (const float*)d_in[3];   // [K]

    const int N = in_sizes[0] / 32;
    const int K = in_sizes[1] / 32;

    float* out = (float*)d_out;

    // workspace carve (bytes): ES[K*32 f32] | counts[K] | hn[K] | eh[K*32 f16] | commit
    char* ws = (char*)d_ws;
    float*    wsES     = (float*)ws;
    float*    wsCounts = (float*)(ws + (size_t)K * 128);
    float*    wsHN     = (float*)(ws + (size_t)K * 132);
    _Float16* wsEH     = (_Float16*)(ws + (size_t)K * 136);
    float*    wsCommit = (float*)(ws + (size_t)K * 200);

    vq_init_kernel<<<dim3(64), dim3(256), 0, stream>>>(
        emb, wsES, wsCounts, wsHN, wsEH, wsCommit, K);

    const size_t smemBytes = (size_t)K * 68 + 1024 + 1024;  // eh + hn + idx + red
    vq_assign_kernel<<<dim3(N / 256), dim3(256), smemBytes, stream>>>(
        z, emb, wsEH, wsHN, wsES, wsCounts, wsCommit, out, N, K);

    vq_finalize_kernel<<<dim3(1), dim3(1024), 0, stream>>>(
        cs, eavg, wsES, wsCounts, wsCommit, out, N, K);
}
